// Lfm2MoeSparseMoeBlock_41205916238390
// MI455X (gfx1250) — matrix operand order repaired and checked by reading, not compiled, not run on hardware
//
#include <hip/hip_runtime.h>
#include <hip/hip_bf16.h>

#define T_TOK 4096
#define D_DIM 1024
#define I_DIM 1024
#define N_EXP 32
#define TOPK  4
#define EPSF  1e-6f
#define M_TILE 64
#define LDH   1032   // LDS row stride in bf16 elements (multiple of 8, conflict-skewed)

typedef __attribute__((ext_vector_type(16))) __bf16        v16bf;
typedef __attribute__((ext_vector_type(8)))  float         v8f;
typedef __attribute__((ext_vector_type(16))) unsigned short v16u;

union BFrag { v16u u; v16bf b; int4 q[2]; };

__device__ __forceinline__ v8f wmma_bf16(v16bf a, v16bf b, v8f c) {
  return __builtin_amdgcn_wmma_f32_16x16x32_bf16(false, a, false, b, (short)0, c,
                                                 false, false);
}

// A fragment (16x32 bf16) from LDS, per ISA layout:
// lane<16: K = k0+0..7 (v0-3), k0+16..23 (v4-7); lane>=16: +8 on both halves.
__device__ __forceinline__ v16bf load_a(const __bf16* lds, int row_base,
                                        int k0, int lane) {
  const int m   = row_base + (lane & 15);
  const int hlf = lane >> 4;
  const __bf16* p = lds + m * LDH + k0 + hlf * 8;
  BFrag f;
  f.q[0] = *(const int4*)(p);
  f.q[1] = *(const int4*)(p + 16);
  return f.b;
}

// B fragment (32x16 bf16) from global fp32 weight rows.
// B column n = weight row (nb+n), ISA K layout per lane:
// lane<16: col=lane, K=k0+0..7, k0+16..23; lane>=16: col=lane-16, +8 on both halves.
__device__ __forceinline__ v16bf load_b(const float* __restrict__ wbase, int ldk,
                                        int k0, int lane) {
  const float* p = wbase + (size_t)(lane & 15) * ldk + k0 + ((lane >> 4) << 3);
  float4 f0 = ((const float4*)p)[0];
  float4 f1 = ((const float4*)p)[1];
  float4 f2 = ((const float4*)p)[4];
  float4 f3 = ((const float4*)p)[5];
  v16bf r;
  r[0]  = (__bf16)f0.x; r[1]  = (__bf16)f0.y; r[2]  = (__bf16)f0.z; r[3]  = (__bf16)f0.w;
  r[4]  = (__bf16)f1.x; r[5]  = (__bf16)f1.y; r[6]  = (__bf16)f1.z; r[7]  = (__bf16)f1.w;
  r[8]  = (__bf16)f2.x; r[9]  = (__bf16)f2.y; r[10] = (__bf16)f2.z; r[11] = (__bf16)f2.w;
  r[12] = (__bf16)f3.x; r[13] = (__bf16)f3.y; r[14] = (__bf16)f3.z; r[15] = (__bf16)f3.w;
  return r;
}

__global__ void zero_kernel(float* __restrict__ out, int n, int* __restrict__ counts) {
  int i = blockIdx.x * blockDim.x + threadIdx.x;
  if (i < n) out[i] = 0.f;
  if (i < N_EXP) counts[i] = 0;
}

// One wave32 per token; lane == expert (E == 32).
__global__ __launch_bounds__(256) void route_kernel(
    const float* __restrict__ x, const float* __restrict__ gw,
    const float* __restrict__ bias, int* __restrict__ counts,
    int* __restrict__ etok, float* __restrict__ ewt) {
  const int lane = threadIdx.x & 31;
  const int t = blockIdx.x * 8 + (threadIdx.x >> 5);

  const float4* xr = (const float4*)(x + (size_t)t * D_DIM);
  const float4* wr = (const float4*)(gw + (size_t)lane * D_DIM);
  float acc = 0.f;
  #pragma unroll 4
  for (int i = 0; i < D_DIM / 4; ++i) {
    float4 a = xr[i], b = wr[i];
    acc += a.x * b.x + a.y * b.y + a.z * b.z + a.w * b.w;
  }
  const float r = 1.f / (1.f + __expf(-acc));   // sigmoid routing weight
  float s = r + bias[lane];                     // selection score

  int selk[TOPK]; float wk[TOPK]; float wsum = 0.f;
  #pragma unroll
  for (int k = 0; k < TOPK; ++k) {
    float bs = s; int bi = lane;
    #pragma unroll
    for (int off = 16; off > 0; off >>= 1) {
      float os = __shfl_xor(bs, off, 32);
      int   oi = __shfl_xor(bi, off, 32);
      if (os > bs || (os == bs && oi < bi)) { bs = os; bi = oi; }
    }
    selk[k] = bi;
    float w = __shfl(r, bi, 32);
    wk[k] = w; wsum += w;
    if (lane == bi) s = -__builtin_inff();
  }
  const float inv = 1.f / (wsum + EPSF);
  if (lane < TOPK) {
    int e = selk[lane];
    int pos = atomicAdd(&counts[e], 1);
    etok[e * T_TOK + pos] = t;
    ewt[e * T_TOK + pos]  = wk[lane] * inv;   // ROUTED_SCALING == 1.0
  }
}

// Block: 64 tokens of one expert, 8 waves; wave w owns N columns [w*128, w*128+128).
__global__ __launch_bounds__(256) void moe_expert_kernel(
    const float* __restrict__ x,      // [T, D]
    const float* __restrict__ gup,    // [E, 2I, D]
    const float* __restrict__ dwn,    // [E, D, I]
    const int*   __restrict__ counts, // [E]
    const int*   __restrict__ etok,   // [E, T]
    const float* __restrict__ ewt,    // [E, T]
    float* __restrict__ out) {        // [T, D]
  extern __shared__ unsigned char smem[];
  __bf16* Xs = (__bf16*)smem;                 // 64 x LDH bf16
  __bf16* Hs = Xs + M_TILE * LDH;             // 64 x LDH bf16
  int*   toks = (int*)(Hs + M_TILE * LDH);    // 64
  float* wts  = (float*)(toks + M_TILE);      // 64

  const int e     = blockIdx.y;
  const int cnt   = counts[e];
  const int start = blockIdx.x * M_TILE;
  if (start >= cnt) return;
  const int nrows = (cnt - start < M_TILE) ? (cnt - start) : M_TILE;

  const int tid = threadIdx.x;
  if (tid < M_TILE) {
    int tk = 0; float w = 0.f;
    if (tid < nrows) {
      tk = etok[e * T_TOK + start + tid];
      w  = ewt[e * T_TOK + start + tid];
    }
    toks[tid] = tk; wts[tid] = w;
  }
  __syncthreads();

  // Stage X rows (fp32 -> bf16) into LDS: 4 threads per row, 256 cols each.
  {
    const int row = tid >> 2;
    const int c0  = (tid & 3) * 256;
    __bf16* dst = Xs + row * LDH + c0;
    if (row < nrows) {
      const float4* src = (const float4*)(x + (size_t)toks[row] * D_DIM + c0);
      #pragma unroll 8
      for (int i = 0; i < 64; ++i) {
        float4 v = src[i];
        dst[i * 4 + 0] = (__bf16)v.x; dst[i * 4 + 1] = (__bf16)v.y;
        dst[i * 4 + 2] = (__bf16)v.z; dst[i * 4 + 3] = (__bf16)v.w;
      }
    } else {
      #pragma unroll 8
      for (int i = 0; i < 256; ++i) dst[i] = (__bf16)0.f;
    }
  }
  __syncthreads();

  const int lane = tid & 31;
  const int wave = tid >> 5;
  const int n0   = wave * 128;
  const int nsel = lane & 15;
  const int hlf  = lane >> 4;

  // ---------- Phase 1: h = silu(X@Wg^T) * (X@Wu^T) ----------
  const float* Wg = gup + (size_t)e * (2 * I_DIM) * D_DIM;
  for (int j = 0; j < 8; ++j) {
    const int nb = n0 + j * 16;
    const float* Bg = Wg + (size_t)nb * D_DIM;
    const float* Bu = Wg + (size_t)(I_DIM + nb) * D_DIM;
    v8f zero = {};
    v8f accg[4] = {zero, zero, zero, zero};
    v8f accu[4] = {zero, zero, zero, zero};
    for (int k0 = 0; k0 < D_DIM; k0 += 32) {
      v16bf bg = load_b(Bg, D_DIM, k0, lane);
      v16bf bu = load_b(Bu, D_DIM, k0, lane);
      if (k0 + 160 < D_DIM) {
        __builtin_prefetch(Bg + (size_t)nsel * D_DIM + k0 + 160, 0, 1);
        __builtin_prefetch(Bu + (size_t)nsel * D_DIM + k0 + 160, 0, 1);
      }
      #pragma unroll
      for (int m = 0; m < 4; ++m) {
        v16bf a = load_a(Xs, m * 16, k0, lane);
        accg[m] = wmma_bf16(a, bg, accg[m]);
        accu[m] = wmma_bf16(a, bu, accu[m]);
      }
    }
    const int col = nb + nsel;
    #pragma unroll
    for (int m = 0; m < 4; ++m) {
      const int rbase = m * 16 + hlf * 8;
      #pragma unroll
      for (int r = 0; r < 8; ++r) {
        float g = accg[m][r], u = accu[m][r];
        float h = (g / (1.f + __expf(-g))) * u;    // silu(g) * u
        Hs[(rbase + r) * LDH + col] = (__bf16)h;
      }
    }
  }
  __syncthreads();

  // ---------- Phase 2: out += (h @ Wd^T) * w, scattered to tokens ----------
  const float* Wd = dwn + (size_t)e * D_DIM * I_DIM;
  for (int j = 0; j < 8; ++j) {
    const int nb = n0 + j * 16;
    const float* Bd = Wd + (size_t)nb * I_DIM;
    v8f zero = {};
    v8f acc[4] = {zero, zero, zero, zero};
    for (int k0 = 0; k0 < I_DIM; k0 += 32) {
      v16bf b = load_b(Bd, I_DIM, k0, lane);
      if (k0 + 160 < I_DIM)
        __builtin_prefetch(Bd + (size_t)nsel * I_DIM + k0 + 160, 0, 1);
      #pragma unroll
      for (int m = 0; m < 4; ++m) {
        v16bf a = load_a(Hs, m * 16, k0, lane);
        acc[m] = wmma_bf16(a, b, acc[m]);
      }
    }
    const int col = nb + nsel;
    #pragma unroll
    for (int m = 0; m < 4; ++m) {
      const int rbase = m * 16 + hlf * 8;
      #pragma unroll
      for (int r = 0; r < 8; ++r) {
        const int row = rbase + r;
        float v = acc[m][r] * wts[row];
        atomicAdd(out + (size_t)toks[row] * D_DIM + col, v);
      }
    }
  }
}

extern "C" void kernel_launch(void* const* d_in, const int* in_sizes, int n_in,
                              void* d_out, int out_size, void* d_ws, size_t ws_size,
                              hipStream_t stream) {
  const float* x    = (const float*)d_in[0];   // hidden_states [T, D]
  const float* gw   = (const float*)d_in[1];   // gate_w [E, D]
  const float* gup  = (const float*)d_in[2];   // gate_up_proj [E, 2I, D]
  const float* dwn  = (const float*)d_in[3];   // down_proj [E, D, I]
  const float* bias = (const float*)d_in[4];   // expert_bias [E]
  float* out = (float*)d_out;

  int*   counts = (int*)d_ws;
  int*   etok   = (int*)((char*)d_ws + 256);
  float* ewt    = (float*)((char*)d_ws + 256 + sizeof(int) * (size_t)N_EXP * T_TOK);

  const int n = T_TOK * D_DIM;
  zero_kernel<<<(n + 255) / 256, 256, 0, stream>>>(out, n, counts);
  route_kernel<<<T_TOK / 8, 256, 0, stream>>>(x, gw, bias, counts, etok, ewt);

  const size_t shmem = (size_t)(2 * M_TILE * LDH) * sizeof(__bf16) +
                       (size_t)M_TILE * (sizeof(int) + sizeof(float));
  moe_expert_kernel<<<dim3(T_TOK / M_TILE, N_EXP), 256, shmem, stream>>>(
      x, gup, dwn, counts, etok, ewt, out);
}
